// PIDEQ_19310172963579
// MI455X (gfx1250) — matrix operand order repaired and checked by reading, not compiled, not run on hardware
//
#include <hip/hip_runtime.h>
#include <math.h>

// ---------------------------------------------------------------------------
// PIDEQ on gfx1250: register-resident DEQ fixed point with f16 WMMA (f32 acc).
//
// GEMM orientation:  D[M=16 states, N=16 batch cols] += A[16x32 f16 = Aw tile]
//                                                    x  B[32x16 f16 = z tile]
// Accumulator (lane = batch col) and B operand (lane = batch col) share the
// lane mapping, so the D->B relayout after tanh is one lane^16 half swap.
// The affine bias (Ab+Bb + Bw x_) is folded into the GEMM as an extra K-tile
// whose B fragment holds the constants (1, x0, x1).
// Each wave owns TWO 16-column batch tiles so every A-fragment ds_load_b128
// pair feeds two WMMAs (1 load : 1 wmma) -> LDS bandwidth balanced vs XDL.
// Both m0/m1 fragments are loaded ahead of the 4 WMMAs of each K-step so the
// scheduler can keep loads in flight behind the matrix ops.
// ---------------------------------------------------------------------------

typedef __attribute__((ext_vector_type(16))) _Float16 v16h;
typedef __attribute__((ext_vector_type(2)))  __fp16   v2h;   // cvt_pkrtz result type
typedef __attribute__((ext_vector_type(8)))  float    v8f;

#define SMAX   200          // true state count
#define SP     224          // padded states (14 tiles of 16 == 7 k-tiles of 32)
#define NT     14           // 16-row state tiles (M direction)
#define KTI    7            // 32-deep K tiles of real states
#define KT2    8            // + 1 bias K-tile
#define NFRAG2 (NT*KT2)     // 112 pre-swizzled A fragments (incl. bias columns)
#define TVAL   10.0f
#define XB0    (-5.0f)
#define XB1    (5.0f)
#define EPSV   1e-4f
#define MAXIT  200

union HalfPack8 { _Float16 h[8]; uint4 u; };
union FragCvt   { uint4 u[2];    v16h  h; };
union NBPack    { v2h p[8];      v16h  v; };

// Branch-free tanh: prefer gfx1250 V_TANH_F32, else exp2/rcp (saturates
// correctly at +-inf; ~1e-6 abs error, far below the f16 state storage).
__device__ __forceinline__ float fast_tanh(float x) {
#if __has_builtin(__builtin_amdgcn_tanhf)
  return __builtin_amdgcn_tanhf(x);
#else
  float e = __builtin_amdgcn_exp2f(x * 2.88539008177793f);  // 2*log2(e)
  return 1.0f - 2.0f * __builtin_amdgcn_rcpf(e + 1.0f);
#endif
}

// tanh on a state-tile pair (D layout) -> next B fragment (lane^16 half swap),
// accumulating convergence norms against the previous fragment.
__device__ __forceinline__ v16h relayout_tanh(const v8f& accm0, const v8f& accm1,
                                              const v16h& oldB, bool low,
                                              float& sumD, float& sumZ) {
  float z0v[8], z1v[8], o0[8], o1[8];
#pragma unroll
  for (int r = 0; r < 8; ++r) { z0v[r] = fast_tanh(accm0[r]); z1v[r] = fast_tanh(accm1[r]); }
#pragma unroll
  for (int r = 0; r < 8; ++r) {
    o0[r] = __shfl_xor(z0v[r], 16, 32);
    o1[r] = __shfl_xor(z1v[r], 16, 32);
  }
  float wlo[8], whi[8];
#pragma unroll
  for (int kk = 0; kk < 8; ++kk) {
    wlo[kk] = low ? z0v[kk] : o1[kk];     // slot kk    (tile-K = kk / 16+kk)
    whi[kk] = low ? o0[kk]  : z1v[kk];    // slot kk+8
    float dlo = wlo[kk] - (float)oldB[kk];
    float dhi = whi[kk] - (float)oldB[kk + 8];
    sumD += dlo * dlo + dhi * dhi;
    sumZ += wlo[kk] * wlo[kk] + whi[kk] * whi[kk];
  }
  NBPack nb;
#pragma unroll
  for (int q = 0; q < 4; ++q) {
    nb.p[q]     = __builtin_amdgcn_cvt_pkrtz(wlo[2 * q], wlo[2 * q + 1]);
    nb.p[q + 4] = __builtin_amdgcn_cvt_pkrtz(whi[2 * q], whi[2 * q + 1]);
  }
  return nb.v;
}

__global__ __launch_bounds__(256) void pideq_kernel(
    const float* __restrict__ T,  const float* __restrict__ X,
    const float* __restrict__ Aw, const float* __restrict__ Ab,
    const float* __restrict__ Bw, const float* __restrict__ Bb,
    const float* __restrict__ Cw, const float* __restrict__ Cb,
    const float* __restrict__ Dw, const float* __restrict__ Db,
    float* __restrict__ Out)
{
  // Weight A-fragments, pre-swizzled: frag = mt*8 + kt, 1 KB each (112 KB).
  __shared__ uint4  sW[NFRAG2 * 64];
  __shared__ float2 sC[SP];       // {Cw[0,:], Cw[1,:]} readout table
  __shared__ float  sRed[2];
  __shared__ int    sCont;

  const int  tid  = threadIdx.x;
  const int  lane = tid & 31;
  const int  wave = tid >> 5;
  const bool low  = (lane < 16);

  // ---- one-time stage: Aw (+ bias columns) -> LDS in A-fragment layout ----
  // A 16x32 f16 fragment: lane l holds row M = l&15;
  // K order chosen here (B uses the same): lanes 0-15: K {0..15}; lanes 16-31: K {16..31}.
  // kt==7 is the bias tile: local K 0,1,2 = {Ab+Bb, Bw[:,0], Bw[:,1]}.
  for (int idx = tid; idx < NFRAG2 * 64; idx += 256) {
    int f  = idx >> 6, rem = idx & 63, ln = rem >> 1, ch = rem & 1;
    int mt = f >> 3,   kt  = f & 7;
    int j  = mt * 16 + (ln & 15);
    int kb = kt * 32 + ((ln < 16) ? (ch ? 8 : 0) : (ch ? 24 : 16));
    HalfPack8 p;
#pragma unroll
    for (int i = 0; i < 8; ++i) {
      int   k = kb + i;
      float v = 0.0f;
      if (j < SMAX) {
        if (k < SP) {
          if (k < SMAX) v = Aw[j * SMAX + k];
        } else {
          int kl = k - SP;
          if      (kl == 0) v = Ab[j] + Bb[j];
          else if (kl == 1) v = Bw[2 * j + 0];
          else if (kl == 2) v = Bw[2 * j + 1];
        }
      }
      p.h[i] = (_Float16)v;
    }
    sW[idx] = p.u;
  }
  for (int j = tid; j < SP; j += 256) {
    float2 ct = make_float2(0.f, 0.f);
    if (j < SMAX) { ct.x = Cw[j]; ct.y = Cw[SMAX + j]; }
    sC[j] = ct;
  }
  if (tid < 2)  sRed[tid] = 0.0f;
  if (tid == 0) sCont = 1;
  __syncthreads();

  // ---- per-lane batch columns (two 16-wide tiles per wave) ---------------
  const int col0 = blockIdx.x * 256 + wave * 32 + (lane & 15);
  const int col1 = col0 + 16;
  float x0a = 2.0f * T[col0] / TVAL - 1.0f;
  float x1a = 2.0f * (X[col0] - XB0) / (XB1 - XB0) - 1.0f;
  float x0b = 2.0f * T[col1] / TVAL - 1.0f;
  float x1b = 2.0f * (X[col1] - XB0) / (XB1 - XB0) - 1.0f;

  // Constant bias B-fragments: low lanes (tile-K 0..15) carry (1, x0, x1).
  v16h biasBa, biasBb;
#pragma unroll
  for (int i = 0; i < 16; ++i) { biasBa[i] = (_Float16)0.0f; biasBb[i] = (_Float16)0.0f; }
  biasBa[0] = (_Float16)(low ? 1.0f : 0.0f);
  biasBa[1] = (_Float16)(low ? x0a  : 0.0f);
  biasBa[2] = (_Float16)(low ? x1a  : 0.0f);
  biasBb[0] = (_Float16)(low ? 1.0f : 0.0f);
  biasBb[1] = (_Float16)(low ? x0b  : 0.0f);
  biasBb[2] = (_Float16)(low ? x1b  : 0.0f);

  v8f zc;
#pragma unroll
  for (int r = 0; r < 8; ++r) zc[r] = 0.0f;

  // z state held as WMMA B fragments (f16, lane = batch col). z0 = 0.
  v16h curBa[KTI], curBb[KTI];
#pragma unroll
  for (int p = 0; p < KTI; ++p)
#pragma unroll
    for (int i = 0; i < 16; ++i) { curBa[p][i] = (_Float16)0.0f; curBb[p][i] = (_Float16)0.0f; }

  // ---- fixed-point iteration ---------------------------------------------
#pragma unroll 1
  for (int it = 0; it < MAXIT; ++it) {
    float sumD = 0.0f, sumZ = 0.0f;
    v16h  nextBa[KTI], nextBb[KTI];

#pragma unroll
    for (int p = 0; p < KTI; ++p) {       // state-tile pair (m0,m1) -> k-tile p
      const int m0 = 2 * p, m1 = 2 * p + 1;
      v8f acc0a = zc, acc1a = zc, acc0b = zc, acc1b = zc;
#pragma unroll
      for (int i = 0; i < KT2; ++i) {     // i==0: bias tile, then 7 state tiles
        const int  kt   = (i == 0) ? 7 : (i - 1);
        const v16h bopA = (i == 0) ? biasBa : curBa[i - 1];
        const v16h bopB = (i == 0) ? biasBb : curBb[i - 1];
        FragCvt a0, a1;
        int b0 = (((m0 << 3) + kt) << 6) + (lane << 1);
        int b1 = (((m1 << 3) + kt) << 6) + (lane << 1);
        a0.u[0] = sW[b0];  a0.u[1] = sW[b0 + 1];   // issue all 4 loads first,
        a1.u[0] = sW[b1];  a1.u[1] = sW[b1 + 1];   // then the 4 matrix ops
        acc0a = __builtin_amdgcn_wmma_f32_16x16x32_f16(
                    false, a0.h, false, bopA, (short)0, acc0a, false, false);
        acc0b = __builtin_amdgcn_wmma_f32_16x16x32_f16(
                    false, a0.h, false, bopB, (short)0, acc0b, false, false);
        acc1a = __builtin_amdgcn_wmma_f32_16x16x32_f16(
                    false, a1.h, false, bopA, (short)0, acc1a, false, false);
        acc1b = __builtin_amdgcn_wmma_f32_16x16x32_f16(
                    false, a1.h, false, bopB, (short)0, acc1b, false, false);
      }
      nextBa[p] = relayout_tanh(acc0a, acc1a, curBa[p], low, sumD, sumZ);
      nextBb[p] = relayout_tanh(acc0b, acc1b, curBb[p], low, sumD, sumZ);
    }

    // workgroup-local convergence: wave reduce -> LDS atomics -> scalar test
#pragma unroll
    for (int off = 16; off >= 1; off >>= 1) {
      sumD += __shfl_xor(sumD, off, 32);
      sumZ += __shfl_xor(sumZ, off, 32);
    }
    if (lane == 0) { atomicAdd(&sRed[0], sumD); atomicAdd(&sRed[1], sumZ); }
    __syncthreads();
    if (tid == 0) {
      float err = sqrtf(sRed[0]) / (sqrtf(sRed[1]) + 1e-9f);
      sCont   = (err > EPSV) ? 1 : 0;
      sRed[0] = 0.0f;  sRed[1] = 0.0f;
    }
#pragma unroll
    for (int p = 0; p < KTI; ++p) { curBa[p] = nextBa[p]; curBb[p] = nextBb[p]; }
    __syncthreads();
    if (!sCont) break;                     // uniform across the block
  }

  // ---- readout: out = C z + Cb + D x_ + Db --------------------------------
  float p0a = 0.0f, p1a = 0.0f, p0b = 0.0f, p1b = 0.0f;
#pragma unroll
  for (int p = 0; p < KTI; ++p) {
#pragma unroll
    for (int kk = 0; kk < 16; ++kk) {
      int    j  = p * 32 + (low ? kk : 16 + kk);
      float2 c  = sC[j];
      float  za = (float)curBa[p][kk];
      float  zb = (float)curBb[p][kk];
      p0a = fmaf(c.x, za, p0a);
      p1a = fmaf(c.y, za, p1a);
      p0b = fmaf(c.x, zb, p0b);
      p1b = fmaf(c.y, zb, p1b);
    }
  }
  p0a += __shfl_xor(p0a, 16, 32);          // combine the two state halves
  p1a += __shfl_xor(p1a, 16, 32);
  p0b += __shfl_xor(p0b, 16, 32);
  p1b += __shfl_xor(p1b, 16, 32);
  if (low) {
    float cb0 = Cb[0], cb1 = Cb[1], db0 = Db[0], db1 = Db[1];
    float d00 = Dw[0], d01 = Dw[1], d10 = Dw[2], d11 = Dw[3];
    Out[col0 * 2 + 0] = p0a + cb0 + d00 * x0a + d01 * x1a + db0;
    Out[col0 * 2 + 1] = p1a + cb1 + d10 * x0a + d11 * x1a + db1;
    Out[col1 * 2 + 0] = p0b + cb0 + d00 * x0b + d01 * x1b + db0;
    Out[col1 * 2 + 1] = p1b + cb1 + d10 * x0b + d11 * x1b + db1;
  }
}

extern "C" void kernel_launch(void* const* d_in, const int* in_sizes, int n_in,
                              void* d_out, int out_size, void* d_ws, size_t ws_size,
                              hipStream_t stream) {
  (void)in_sizes; (void)n_in; (void)out_size; (void)d_ws; (void)ws_size;
  const float* T  = (const float*)d_in[0];
  const float* X  = (const float*)d_in[1];
  const float* Aw = (const float*)d_in[2];
  const float* Ab = (const float*)d_in[3];
  const float* Bw = (const float*)d_in[4];
  const float* Bb = (const float*)d_in[5];
  const float* Cw = (const float*)d_in[6];
  const float* Cb = (const float*)d_in[7];
  const float* Dw = (const float*)d_in[8];
  const float* Db = (const float*)d_in[9];
  float* Out = (float*)d_out;

  const int BATCH = 131072;
  dim3 grid(BATCH / 256), block(256);
  pideq_kernel<<<grid, block, 0, stream>>>(T, X, Aw, Ab, Bw, Bb, Cw, Cb, Dw, Db, Out);
}
